// RVMixtureSynthesizers_3418793968314
// MI455X (gfx1250) — matrix operand order repaired and checked by reading, not compiled, not run on hardware
//
#include <hip/hip_runtime.h>

// ---------------------------------------------------------------------------
// Attention block for MI455X (gfx1250, wave32).
//   q = query @ Wq^T + bq ; k = key @ Wk^T + bk     (f16 into ws, inline f32->f16)
//   E = q @ k^T + attn_bias                          (f32 into d_out attn region)
//   A = softmax(E) in place  (+ f16 copy into ws for the AV GEMM)
//   out = A @ value                                  (f32 into d_out out region)
//
// GEMMs: v_wmma_f32_16x16x32_f16, 128x128x32 tiles, 8 waves/block.
// energy/av use GLOBAL_LOAD_ASYNC_TO_LDS_B128 + s_wait_asynccnt with
// double-buffered LDS (CDNA5 async-to-LDS path, ASYNCcnt-tracked).
// d_out = [ out : B*S*D floats | attention : B*S*S floats ]
// ---------------------------------------------------------------------------

typedef _Float16 f16;
typedef _Float16 v16h __attribute__((ext_vector_type(16)));
typedef _Float16 v8h  __attribute__((ext_vector_type(8)));
typedef float    v8f  __attribute__((ext_vector_type(8)));

#define BB 16
#define SS 1024
#define DD 1024

#define BM 128
#define BN 128
#define BK 32
#define NTHREADS 256      // 8 waves of 32
#define NKSTEPS (DD / BK) // 32

// ---- WMMA fragment loads (contiguous k = 16*(lane/16) + s, same order for A and B) --------------------

// A tile in LDS: [BM][BK] f16, m-major. 16x32 A frag:
// lane l (l<16): row M=l, K=0..15; lane l+16: row M=l-16, K=16..31.
__device__ __forceinline__ v16h load_a_frag(const f16* a_tile, int row0, int lane) {
    const int row   = row0 + (lane & 15);
    const int khalf = (lane >> 4) * 16;
    const f16* p = a_tile + row * BK;
    return *(const v16h*)(p + khalf);
}

// B tile in LDS: [BN][BK] f16, n-major. 32x16 B frag:
// lane l (l<16): col N=l, K=0..15; lane l+16: col N=l-16, K=16..31 -> one 32B read.
__device__ __forceinline__ v16h load_b_frag(const f16* b_tile, int col0, int lane) {
    const int n     = col0 + (lane & 15);
    const int kbase = (lane >> 4) * 16;
    return *(const v16h*)(b_tile + n * BK + kbase);
}

__device__ __forceinline__ v8f wmma_f16(v16h a, v16h b, v8f c) {
    return __builtin_amdgcn_wmma_f32_16x16x32_f16(false, a, false, b, (short)0, c, false, false);
}

// ---- Async global->LDS tile copy (CDNA5 GLOBAL_LOAD_ASYNC_TO_LDS_B128) -----
// Copies a [rows=BM/BN][BK] f16 tile whose global rows are contiguous (64B).
// 4096 halves = 512 x 16B chunks; 256 threads -> 2 async b128 ops per thread.
__device__ __forceinline__ void async_tile_f16(const f16* __restrict__ gbase, int ld,
                                               int kk0, f16* lbase, int tid) {
    #pragma unroll
    for (int i = 0; i < 2; ++i) {
        const int c   = tid + i * NTHREADS;
        const int row = c >> 2;             // 4 chunks per 32-half row
        const int col = (c & 3) * 8;        // halves
        const f16* g = gbase + (size_t)row * ld + kk0 + col;
        const unsigned l = (unsigned)(unsigned long long)(lbase + row * BK + col);
        asm volatile("global_load_async_to_lds_b128 %0, %1, off"
                     :: "v"(l), "v"(g) : "memory");
    }
}

#define WAIT_ASYNC(n) asm volatile("s_wait_asynccnt " #n ::: "memory")

// ---- Stage 1: projection  Y_f16[m, n] = sum_d X[m, d] * W[n, d] + bias[n] --
// f32 sources converted inline during the LDS fill (reads f32 exactly once;
// a pre-convert pass would add an extra full read+write of X).
__global__ __launch_bounds__(NTHREADS)
void proj_kernel(const float* __restrict__ X, const float* __restrict__ W,
                 const float* __restrict__ bias, f16* __restrict__ Y) {
    __shared__ __align__(16) f16 As[BM * BK];
    __shared__ __align__(16) f16 Bs[BN * BK];

    const int m0  = blockIdx.x * BM;       // over B*S rows
    const int n0  = blockIdx.y * BN;       // over D cols
    const int tid = threadIdx.x;
    const int lane = tid & 31;
    const int wid  = tid >> 5;
    const int wm = (wid >> 1) * 32;        // 4 waves in M, 32 rows each
    const int wn = (wid & 1) * 64;         // 2 waves in N, 64 cols each

    v8f acc[2][4] = {};

    for (int kk0 = 0; kk0 < DD; kk0 += BK) {
        #pragma unroll
        for (int i = tid; i < BM * BK; i += NTHREADS) {
            const int r = i / BK, c = i % BK;
            As[r * BK + c] = (f16)X[(size_t)(m0 + r) * DD + kk0 + c];
            Bs[r * BK + c] = (f16)W[(size_t)(n0 + r) * DD + kk0 + c];  // W row-major [n][d]
        }
        __syncthreads();

        v16h afr[2], bfr[4];
        #pragma unroll
        for (int mt = 0; mt < 2; ++mt) afr[mt] = load_a_frag(As, wm + mt * 16, lane);
        #pragma unroll
        for (int nt = 0; nt < 4; ++nt) bfr[nt] = load_b_frag(Bs, wn + nt * 16, lane);
        #pragma unroll
        for (int mt = 0; mt < 2; ++mt)
            #pragma unroll
            for (int nt = 0; nt < 4; ++nt)
                acc[mt][nt] = wmma_f16(afr[mt], bfr[nt], acc[mt][nt]);
        __syncthreads();
    }

    #pragma unroll
    for (int mt = 0; mt < 2; ++mt)
        #pragma unroll
        for (int nt = 0; nt < 4; ++nt)
            #pragma unroll
            for (int i = 0; i < 8; ++i) {
                const int row = m0 + wm + mt * 16 + (lane >> 4) * 8 + i;
                const int col = n0 + wn + nt * 16 + (lane & 15);
                Y[(size_t)row * DD + col] = (f16)(acc[mt][nt][i] + bias[col]);
            }
}

// ---- Stage 2: energy  E[s, t] = sum_d q[s,d]*k[t,d] + attn_bias[s,t] -------
// Both operands f16 in ws; fully async double-buffered global->LDS.
__global__ __launch_bounds__(NTHREADS)
void energy_kernel(const f16* __restrict__ Q, const f16* __restrict__ Kp,
                   const float* __restrict__ bias2d, float* __restrict__ E) {
    __shared__ __align__(16) f16 As[2][BM * BK];
    __shared__ __align__(16) f16 Bs[2][BN * BK];

    const int b = blockIdx.z;
    Q  += (size_t)b * SS * DD;
    Kp += (size_t)b * SS * DD;
    E  += (size_t)b * SS * SS;

    const int m0  = blockIdx.x * BM;       // s
    const int n0  = blockIdx.y * BN;       // t
    const int tid = threadIdx.x;
    const int lane = tid & 31;
    const int wid  = tid >> 5;
    const int wm = (wid >> 1) * 32;
    const int wn = (wid & 1) * 64;

    const f16* Arow = Q  + (size_t)m0 * DD;
    const f16* Brow = Kp + (size_t)n0 * DD;

    v8f acc[2][4] = {};

    // Prologue: fill buffer 0 (4 async b128 per thread, 2 per tile).
    async_tile_f16(Arow, DD, 0, &As[0][0], tid);
    async_tile_f16(Brow, DD, 0, &Bs[0][0], tid);

    for (int k = 0; k < NKSTEPS; ++k) {
        const int cur = k & 1, nxt = cur ^ 1;
        if (k + 1 < NKSTEPS) {
            async_tile_f16(Arow, DD, (k + 1) * BK, &As[nxt][0], tid);
            async_tile_f16(Brow, DD, (k + 1) * BK, &Bs[nxt][0], tid);
            WAIT_ASYNC(8);     // allows 8 in flight = 4 just-issued (nxt) + 4 cur; cur not awaited
        } else {
            WAIT_ASYNC(0);
        }
        __syncthreads();       // barrier only; cur async fills not awaited

        v16h afr[2], bfr[4];
        #pragma unroll
        for (int mt = 0; mt < 2; ++mt) afr[mt] = load_a_frag(&As[cur][0], wm + mt * 16, lane);
        #pragma unroll
        for (int nt = 0; nt < 4; ++nt) bfr[nt] = load_b_frag(&Bs[cur][0], wn + nt * 16, lane);
        #pragma unroll
        for (int mt = 0; mt < 2; ++mt)
            #pragma unroll
            for (int nt = 0; nt < 4; ++nt)
                acc[mt][nt] = wmma_f16(afr[mt], bfr[nt], acc[mt][nt]);
        __syncthreads();       // cur reads done; reusable next iteration
    }

    #pragma unroll
    for (int mt = 0; mt < 2; ++mt)
        #pragma unroll
        for (int nt = 0; nt < 4; ++nt)
            #pragma unroll
            for (int i = 0; i < 8; ++i) {
                const int s = m0 + wm + mt * 16 + (lane >> 4) * 8 + i;
                const int t = n0 + wn + nt * 16 + (lane & 15);
                E[(size_t)s * SS + t] = acc[mt][nt][i] + bias2d[(size_t)s * SS + t];
            }
}

// ---- Stage 3: row softmax in place; also emit f16 copy for AV GEMM ---------
__global__ __launch_bounds__(NTHREADS)
void softmax_kernel(float* __restrict__ attn, f16* __restrict__ attn16) {
    __shared__ float red[NTHREADS];
    const size_t rowoff = (size_t)blockIdx.x * SS;
    float* p  = attn   + rowoff;
    f16*   p6 = attn16 + rowoff;
    const int tid = threadIdx.x;

    float x[4];
    float m = -__builtin_inff();
    #pragma unroll
    for (int j = 0; j < 4; ++j) { x[j] = p[j * NTHREADS + tid]; m = fmaxf(m, x[j]); }

    red[tid] = m; __syncthreads();
    #pragma unroll
    for (int s = NTHREADS / 2; s > 0; s >>= 1) {
        if (tid < s) red[tid] = fmaxf(red[tid], red[tid + s]);
        __syncthreads();
    }
    m = red[0]; __syncthreads();

    float sum = 0.f;
    #pragma unroll
    for (int j = 0; j < 4; ++j) { x[j] = __expf(x[j] - m); sum += x[j]; }

    red[tid] = sum; __syncthreads();
    #pragma unroll
    for (int s = NTHREADS / 2; s > 0; s >>= 1) {
        if (tid < s) red[tid] += red[tid + s];
        __syncthreads();
    }
    const float inv = 1.0f / red[0];
    #pragma unroll
    for (int j = 0; j < 4; ++j) {
        const float v = x[j] * inv;
        p [j * NTHREADS + tid] = v;
        p6[j * NTHREADS + tid] = (f16)v;
    }
}

// ---- Stage 4: out[s, d] = sum_t attn[s,t] * value[t,d] ---------------------
// A (attn16) async double-buffered; V converted f32->f16 + transposed into
// n-major LDS manually (global layout mismatches LDS layout), double-buffered.
__global__ __launch_bounds__(NTHREADS)
void av_kernel(const f16* __restrict__ A16, const float* __restrict__ V,
               float* __restrict__ O) {
    __shared__ __align__(16) f16 As[2][BM * BK];
    __shared__ __align__(16) f16 Bs[2][BN * BK];

    const int b = blockIdx.z;
    A16 += (size_t)b * SS * SS;
    V   += (size_t)b * SS * DD;
    O   += (size_t)b * SS * DD;

    const int m0  = blockIdx.x * BM;       // s
    const int n0  = blockIdx.y * BN;       // d
    const int tid = threadIdx.x;
    const int lane = tid & 31;
    const int wid  = tid >> 5;
    const int wm = (wid >> 1) * 32;
    const int wn = (wid & 1) * 64;

    const f16* Arow = A16 + (size_t)m0 * SS;

    v8f acc[2][4] = {};

    // Prologue: buffer 0.
    async_tile_f16(Arow, SS, 0, &As[0][0], tid);
    #pragma unroll
    for (int i = tid; i < BN * BK; i += NTHREADS) {
        const int n = i % BN, c = i / BN;                    // coalesced over d(=n)
        Bs[0][n * BK + c] = (f16)V[(size_t)c * DD + n0 + n];
    }

    for (int k = 0; k < NKSTEPS; ++k) {     // K dim = t
        const int cur = k & 1, nxt = cur ^ 1;
        if (k + 1 < NKSTEPS) {
            const int kk1 = (k + 1) * BK;
            async_tile_f16(Arow, SS, kk1, &As[nxt][0], tid);
            #pragma unroll
            for (int i = tid; i < BN * BK; i += NTHREADS) {
                const int n = i % BN, c = i / BN;
                Bs[nxt][n * BK + c] = (f16)V[(size_t)(kk1 + c) * DD + n0 + n];
            }
            WAIT_ASYNC(2);    // 2 just-issued A chunks in flight; cur A done
        } else {
            WAIT_ASYNC(0);
        }
        __syncthreads();       // cur A + cur B (ds stores drained) visible

        v16h afr[2], bfr[4];
        #pragma unroll
        for (int mt = 0; mt < 2; ++mt) afr[mt] = load_a_frag(&As[cur][0], wm + mt * 16, lane);
        #pragma unroll
        for (int nt = 0; nt < 4; ++nt) bfr[nt] = load_b_frag(&Bs[cur][0], wn + nt * 16, lane);
        #pragma unroll
        for (int mt = 0; mt < 2; ++mt)
            #pragma unroll
            for (int nt = 0; nt < 4; ++nt)
                acc[mt][nt] = wmma_f16(afr[mt], bfr[nt], acc[mt][nt]);
        __syncthreads();
    }

    #pragma unroll
    for (int mt = 0; mt < 2; ++mt)
        #pragma unroll
        for (int nt = 0; nt < 4; ++nt)
            #pragma unroll
            for (int i = 0; i < 8; ++i) {
                const int s = m0 + wm + mt * 16 + (lane >> 4) * 8 + i;
                const int d = n0 + wn + nt * 16 + (lane & 15);
                O[(size_t)s * DD + d] = acc[mt][nt][i];
            }
}

// ---------------------------------------------------------------------------
extern "C" void kernel_launch(void* const* d_in, const int* in_sizes, int n_in,
                              void* d_out, int out_size, void* d_ws, size_t ws_size,
                              hipStream_t stream) {
    const float* query = (const float*)d_in[0];
    const float* key   = (const float*)d_in[1];
    const float* value = (const float*)d_in[2];
    const float* bias2 = (const float*)d_in[3];   // [1,S,S]
    const float* Wq    = (const float*)d_in[4];
    const float* bq    = (const float*)d_in[5];
    const float* Wk    = (const float*)d_in[6];
    const float* bk    = (const float*)d_in[7];

    float* out_o = (float*)d_out;                             // [B,S,D]
    float* out_a = out_o + (size_t)BB * SS * DD;              // [B,S,S]

    // ws (64 MB): region0 = q16 (dead after energy, reused as attn16); region1 = k16.
    f16* qf16   = (f16*)d_ws;
    f16* kf16   = qf16 + (size_t)BB * SS * DD;
    f16* attn16 = qf16;   // reuse region0 after energy

    // Stage 1: projections over flattened [B*S, D] rows.
    {
        dim3 grid((BB * SS) / BM, DD / BN, 1);
        proj_kernel<<<grid, NTHREADS, 0, stream>>>(query, Wq, bq, qf16);
        proj_kernel<<<grid, NTHREADS, 0, stream>>>(key,   Wk, bk, kf16);
    }
    // Stage 2: energy -> attention region of d_out (async double-buffered).
    {
        dim3 grid(SS / BM, SS / BN, BB);
        energy_kernel<<<grid, NTHREADS, 0, stream>>>(qf16, kf16, bias2, out_a);
    }
    // Stage 3: softmax in place + f16 copy for AV.
    softmax_kernel<<<dim3(BB * SS), NTHREADS, 0, stream>>>(out_a, attn16);
    // Stage 4: out = attn @ value (async A, double-buffered).
    {
        dim3 grid(SS / BM, DD / BN, BB);
        av_kernel<<<grid, NTHREADS, 0, stream>>>(attn16, value, out_o);
    }
}
